// Stage1FlowNet_84524956385322
// MI455X (gfx1250) — matrix operand order repaired and checked by reading, not compiled, not run on hardware
//
#include <hip/hip_runtime.h>
#include <math.h>
#include <stdint.h>

// ---------------- problem constants (match reference) ----------------
#define HD     128                 // hidden C
#define NQTOT  (8*2048)            // B*NQ = 16384 query nodes
#define NATOT  (8*64)              // B*NA = 512 atom nodes
#define EAQN   131072
#define EQQN   262144
#define MIK    385                 // 3C+1 message input dim (sq moved to col 384)
#define MIKP   416                 // padded to 32
#define MILD   448                 // LDS row stride (16B-aligned)

typedef __attribute__((ext_vector_type(16))) _Float16 v16h;
typedef __attribute__((ext_vector_type(8)))  _Float16 v8h;
typedef __attribute__((ext_vector_type(4)))  _Float16 v4h;
typedef __attribute__((ext_vector_type(8)))  float    v8f;
typedef __attribute__((address_space(3)))    _Float16 f16_lds;

// fast silu: one v_exp + one v_rcp (no IEEE div chain)
__device__ __forceinline__ float dsilu(float x) {
  return x * __builtin_amdgcn_rcpf(1.0f + __expf(-x));
}

// Build a v16h fragment from two 16-byte-aligned 8-half chunks (b128 loads).
__device__ __forceinline__ v16h frag2(const _Float16* p0, const _Float16* p1) {
  v8h a = *reinterpret_cast<const v8h*>(p0);
  v8h b = *reinterpret_cast<const v8h*>(p1);
  v16h r;
#pragma unroll
  for (int i = 0; i < 8; ++i) { r[i] = a[i]; r[i + 8] = b[i]; }
  return r;
}

// Two 16x16 output tiles sharing each B fragment:
//   acc0 += A[0:16) x W^T ;  acc1 += A[16:32) x W^T
__device__ __forceinline__ void wmma_pair(const _Float16* A, int lda,
                                          const _Float16* Wt, int kp,
                                          int nBase, int K, int lane,
                                          v8f& acc0, v8f& acc1) {
  const int hf = lane >> 4, l = lane & 15;
  const _Float16* ar0 = A + l * lda + hf * 8;
  const _Float16* ar1 = ar0 + 16 * lda;
  const _Float16* br  = Wt + (long)(nBase + l) * kp + hf * 8;
  for (int k0 = 0; k0 < K; k0 += 32) {
    v16h b  = frag2(br + k0, br + k0 + 16);
    v16h a0 = frag2(ar0 + k0, ar0 + k0 + 16);
    acc0 = __builtin_amdgcn_wmma_f32_16x16x32_f16(false, a0, false, b,
                                                  (short)0, acc0, false, false);
    v16h a1 = frag2(ar1 + k0, ar1 + k0 + 16);
    acc1 = __builtin_amdgcn_wmma_f32_16x16x32_f16(false, a1, false, b,
                                                  (short)0, acc1, false, false);
  }
}

// async row-segment copy: 16 bytes per lane, global -> LDS, ASYNCcnt-tracked
__device__ __forceinline__ void async_b128(unsigned lds_addr, uint64_t gaddr) {
  asm volatile("global_load_async_to_lds_b128 %0, %1, off"
               :: "v"(lds_addr), "v"(gaddr) : "memory");
}
__device__ __forceinline__ void wait_async0() {
  asm volatile("s_wait_asynccnt 0x0" ::: "memory");
}

// ---------------- weight prep: fp32 [K][N] -> f16 [N][Kp] (zero-padded) ----
__global__ void transpose_f16_kernel(const float* __restrict__ W,
                                     _Float16* __restrict__ Wt,
                                     int K, int N, int Kp) {
  int idx = blockIdx.x * blockDim.x + threadIdx.x;
  if (idx >= N * Kp) return;
  int n = idx / Kp, k = idx - n * Kp;
  Wt[idx] = (k < K) ? (_Float16)W[(long)k * N + n] : (_Float16)0.0f;
}

// msg_W1 transpose with row permutation so mi layout is
// [h_src(0:128) | h_tgt(128:256) | t_emb(256:384) | sq(384)] (16B-aligned segs)
// reference rows: 0:128 h_src, 128:256 h_tgt, 256 sq, 257:385 t_emb
__global__ void transpose_msgW1_kernel(const float* __restrict__ W,   // [385][256]
                                       _Float16* __restrict__ Wt) {   // [256][416]
  int idx = blockIdx.x * blockDim.x + threadIdx.x;
  if (idx >= 256 * MIKP) return;
  int n = idx / MIKP, kp = idx - n * MIKP;
  int k = (kp < 256) ? kp : (kp < 384 ? kp + 1 : (kp == 384 ? 256 : -1));
  Wt[idx] = (k >= 0) ? (_Float16)W[(long)k * 256 + n] : (_Float16)0.0f;
}

// ---------------- atom projection -> f16 mirror (gather-only consumer) ----
__global__ void atom_proj_kernel(const float* __restrict__ feat,
                                 const float* __restrict__ W,
                                 const float* __restrict__ b,
                                 _Float16* __restrict__ h_atom16) {
  int idx = blockIdx.x * blockDim.x + threadIdx.x;
  if (idx >= NATOT * HD) return;
  int i = idx / HD, c = idx - i * HD;
  float acc = b[c];
  for (int k = 0; k < 256; ++k) acc += feat[(long)i * 256 + k] * W[(long)k * HD + c];
  h_atom16[idx] = (_Float16)acc;
}

// ---------------- time embedding MLP (WMMA) -> f16 mirror ------------------
__global__ __launch_bounds__(32)
void time_kernel(const float* __restrict__ tq,
                 const _Float16* __restrict__ W1t, const float* __restrict__ b1,
                 const _Float16* __restrict__ W2t, const float* __restrict__ b2,
                 _Float16* __restrict__ temb16) {
  __shared__ alignas(16) _Float16 s_emb[32][HD];    //  8 KB
  __shared__ alignas(16) _Float16 s_hid[32][512];   // 32 KB
  __shared__ float s_t[32];
  const int lane = threadIdx.x & 31;
  const int hf = lane >> 4, l = lane & 15;
  const int node0 = blockIdx.x * 32;

  s_t[lane] = tq[node0 + lane];
  __syncthreads();
  for (int m = 0; m < 32; ++m) {
    float t = s_t[m];
    for (int c = lane; c < HD; c += 32) {
      int f = c & 63;
      float arg = t * __expf(-9.210340371976184f * (float)f * (1.0f / 63.0f));
      s_emb[m][c] = (_Float16)((c < 64) ? __sinf(arg) : __cosf(arg));
    }
  }
  __syncthreads();
  for (int nt = 0; nt < 32; ++nt) {          // N = 512, K = 128
    v8f a0 = {}, a1 = {};
    wmma_pair(&s_emb[0][0], HD, W1t, HD, nt * 16, HD, lane, a0, a1);
    float bb = b1[nt * 16 + l];
#pragma unroll
    for (int r = 0; r < 8; ++r) {
      int m = r + hf * 8;
      s_hid[m][nt * 16 + l]      = (_Float16)dsilu(a0[r] + bb);
      s_hid[m + 16][nt * 16 + l] = (_Float16)dsilu(a1[r] + bb);
    }
  }
  __syncthreads();
  for (int nt = 0; nt < 8; ++nt) {           // N = 128, K = 512
    v8f a0 = {}, a1 = {};
    wmma_pair(&s_hid[0][0], 512, W2t, 512, nt * 16, 512, lane, a0, a1);
    float bb = b2[nt * 16 + l];
    int n = nt * 16 + l;
#pragma unroll
    for (int r = 0; r < 8; ++r) {
      int m = r + hf * 8;
      temb16[(long)(node0 + m) * HD + n]      = (_Float16)(a0[r] + bb);
      temb16[(long)(node0 + m + 16) * HD + n] = (_Float16)(a1[r] + bb);
    }
  }
}

// ---------------- init query features ----------------
__global__ void qdist_kernel(const float* __restrict__ x,
                             const float* __restrict__ ap,
                             float* __restrict__ ld) {
  int q = blockIdx.x * blockDim.x + threadIdx.x;
  if (q >= NQTOT) return;
  float qx = x[q * 3 + 0], qy = x[q * 3 + 1], qz = x[q * 3 + 2];
  const float* a = ap + (long)(q >> 11) * 64 * 3;
  float s = 0.0f;
  for (int i = 0; i < 64; ++i) {
    float dx = qx - a[i * 3 + 0], dy = qy - a[i * 3 + 1], dz = qz - a[i * 3 + 2];
    s += dx * dx + dy * dy + dz * dz;
  }
  ld[q] = log1pf(sqrtf(s * (1.0f / 64.0f)) * 0.125f);
}

__global__ void qhid_kernel(const float* __restrict__ ld,
                            const float* __restrict__ W1, const float* __restrict__ b1,
                            float* __restrict__ qh) {
  int idx = blockIdx.x * blockDim.x + threadIdx.x;
  if (idx >= NQTOT * HD) return;
  int q = idx >> 7, c = idx & (HD - 1);
  qh[idx] = dsilu(ld[q] * W1[c] + b1[c]);
}

// out[idx] = b2[n] + sum_c src[q][c]*W2[c][n]; also writes f16 mirror
__global__ void dense128_kernel(const float* __restrict__ src,
                                const float* __restrict__ W2, const float* __restrict__ b2,
                                float* __restrict__ out, _Float16* __restrict__ out16) {
  int idx = blockIdx.x * blockDim.x + threadIdx.x;
  if (idx >= NQTOT * HD) return;
  int q = idx >> 7, n = idx & (HD - 1);
  const float* s = src + (long)q * HD;
  float acc = b2[n];
  for (int c = 0; c < HD; ++c) acc += s[c] * W2[(long)c * HD + n];
  out[idx] = acc;
  out16[idx] = (_Float16)acc;
}

// ---------------- EGNN edge pass (hot kernel) ----------------
// One wave per 32-edge tile; async-DMA gather (f16 rows) into LDS;
// two 16-row WMMA tiles share every weight B fragment.
__global__ __launch_bounds__(32)
void edge_kernel(const int* __restrict__ esrc, const int* __restrict__ etgt,
                 const _Float16* __restrict__ h16src, const _Float16* __restrict__ h16tgt,
                 const float* __restrict__ psrc, const float* __restrict__ ptgt,
                 const _Float16* __restrict__ temb16,
                 const _Float16* __restrict__ W1t, const float* __restrict__ b1,
                 const _Float16* __restrict__ W2t, const float* __restrict__ b2,
                 const _Float16* __restrict__ cW1t, const float* __restrict__ cb1,
                 const float* __restrict__ cW2, const float* __restrict__ cb2,
                 float* __restrict__ agg, float* __restrict__ vel) {
  __shared__ alignas(16) _Float16 s_mi [32][MILD];   // 28 KB (also cw scratch)
  __shared__ alignas(16) _Float16 s_hid[32][256];    // 16 KB
  __shared__ alignas(16) _Float16 s_msg[32][HD];     //  8 KB

  const int lane = threadIdx.x & 31;
  const int hf = lane >> 4, l = lane & 15;
  const int e0 = blockIdx.x * 32;

  // phase 1: this lane's edge -> registers (live across all phases; 1 wave/block)
  const int ei = e0 + lane;
  const int sIdx = esrc[ei], tIdx = etgt[ei];
  float rx = ptgt[(long)tIdx * 3 + 0] - psrc[(long)sIdx * 3 + 0];
  float ry = ptgt[(long)tIdx * 3 + 1] - psrc[(long)sIdx * 3 + 1];
  float rz = ptgt[(long)tIdx * 3 + 2] - psrc[(long)sIdx * 3 + 2];
  s_mi[lane][384] = (_Float16)(rx * rx + ry * ry + rz * rz);   // sq column

  // phase 2: async DMA gather: 3 segments x 32 rows x 256B, 16B per lane.
  // 2 rows per issue: row r = 2*ep + hf, byte chunk = (lane&15)*16
  {
    const unsigned miBase = (unsigned)(unsigned long long)(f16_lds*)&s_mi[0][0];
    const unsigned chunk = (unsigned)l * 16u;
    const uint64_t gsrc = (uint64_t)h16src;
    const uint64_t gtgt = (uint64_t)h16tgt;
    const uint64_t gemb = (uint64_t)temb16;
    for (int ep = 0; ep < 16; ++ep) {
      int r = ep * 2 + hf;
      int s = __shfl(sIdx, r, 32);
      int t = __shfl(tIdx, r, 32);
      unsigned ldsRow = miBase + (unsigned)r * (MILD * 2);
      async_b128(ldsRow + chunk,        gsrc + (uint64_t)(unsigned)s * 256u + chunk);
      async_b128(ldsRow + 256u + chunk, gtgt + (uint64_t)(unsigned)t * 256u + chunk);
      async_b128(ldsRow + 512u + chunk, gemb + (uint64_t)(unsigned)t * 256u + chunk);
    }
    for (int c = MIK + lane; c < MIKP; c += 32)
#pragma unroll
      for (int e = 0; e < 32; ++e) s_mi[e][c] = (_Float16)0.0f;
    wait_async0();
  }
  __syncthreads();

  // phase 3: message layer 1 (416 -> 256), silu
  for (int nt = 0; nt < 16; ++nt) {
    v8f a0 = {}, a1 = {};
    wmma_pair(&s_mi[0][0], MILD, W1t, MIKP, nt * 16, MIKP, lane, a0, a1);
    float bb = b1[nt * 16 + l];
#pragma unroll
    for (int r = 0; r < 8; ++r) {
      int m = r + hf * 8;
      s_hid[m][nt * 16 + l]      = (_Float16)dsilu(a0[r] + bb);
      s_hid[m + 16][nt * 16 + l] = (_Float16)dsilu(a1[r] + bb);
    }
  }
  __syncthreads();

  // phase 4: message layer 2 (256 -> 128), silu; scatter agg (per-row tgt via shfl)
  for (int nt = 0; nt < 8; ++nt) {
    v8f a0 = {}, a1 = {};
    wmma_pair(&s_hid[0][0], 256, W2t, 256, nt * 16, 256, lane, a0, a1);
    float bb = b2[nt * 16 + l];
    int n = nt * 16 + l;
#pragma unroll
    for (int r = 0; r < 8; ++r) {
      int m = r + hf * 8;
      float v0 = dsilu(a0[r] + bb);
      float v1 = dsilu(a1[r] + bb);
      s_msg[m][n]      = (_Float16)v0;
      s_msg[m + 16][n] = (_Float16)v1;
      int t0 = __shfl(tIdx, m, 32);
      int t1 = __shfl(tIdx, m + 16, 32);
      atomicAdd(&agg[(long)t0 * HD + n], v0);
      atomicAdd(&agg[(long)t1 * HD + n], v1);
    }
  }
  __syncthreads();

  // phase 5: coord-weight layer 1 (128 -> 64), silu (reuse s_mi as scratch)
  for (int nt = 0; nt < 4; ++nt) {
    v8f a0 = {}, a1 = {};
    wmma_pair(&s_msg[0][0], HD, cW1t, HD, nt * 16, HD, lane, a0, a1);
    float bb = cb1[nt * 16 + l];
#pragma unroll
    for (int r = 0; r < 8; ++r) {
      int m = r + hf * 8;
      s_mi[m][nt * 16 + l]      = (_Float16)dsilu(a0[r] + bb);
      s_mi[m + 16][nt * 16 + l] = (_Float16)dsilu(a1[r] + bb);
    }
  }
  __syncthreads();

  // phase 6: 64 -> 1, tanh, scatter vel (own edge per lane, rel still in regs)
  {
    float acc = cb2[0];
    for (int c = 0; c < 64; ++c) acc += (float)s_mi[lane][c] * cW2[c];
    float wt = tanhf(acc);
    atomicAdd(&vel[(long)tIdx * 3 + 0], wt * rx);
    atomicAdd(&vel[(long)tIdx * 3 + 1], wt * ry);
    atomicAdd(&vel[(long)tIdx * 3 + 2], wt * rz);
  }
}

// ---------------- node update: h = LN(h + MLP([h|agg])); writes f16 mirror --
__global__ __launch_bounds__(32)
void node_kernel(float* __restrict__ h, _Float16* __restrict__ h16,
                 const float* __restrict__ agg,
                 const _Float16* __restrict__ W1t, const float* __restrict__ b1,
                 const _Float16* __restrict__ W2t, const float* __restrict__ b2,
                 const float* __restrict__ lng, const float* __restrict__ lnb) {
  __shared__ alignas(16) _Float16 s_u  [32][256];   // 16 KB
  __shared__ alignas(16) _Float16 s_h2 [32][HD];    //  8 KB
  __shared__ float                s_out[32][HD];    // 16 KB
  const int lane = threadIdx.x & 31;
  const int hf = lane >> 4, l = lane & 15;
  const int node0 = blockIdx.x * 32;

  {
    const int c0 = lane * 4;
    for (int m = 0; m < 32; ++m) {
      float4 vh = *(const float4*)(h   + (long)(node0 + m) * HD + c0);
      float4 va = *(const float4*)(agg + (long)(node0 + m) * HD + c0);
      v4h hh = { (_Float16)vh.x, (_Float16)vh.y, (_Float16)vh.z, (_Float16)vh.w };
      v4h ha = { (_Float16)va.x, (_Float16)va.y, (_Float16)va.z, (_Float16)va.w };
      *(v4h*)&s_u[m][c0]      = hh;
      *(v4h*)&s_u[m][HD + c0] = ha;
    }
  }
  __syncthreads();
  for (int nt = 0; nt < 8; ++nt) {           // 256 -> 128, silu
    v8f a0 = {}, a1 = {};
    wmma_pair(&s_u[0][0], 256, W1t, 256, nt * 16, 256, lane, a0, a1);
    float bb = b1[nt * 16 + l];
#pragma unroll
    for (int r = 0; r < 8; ++r) {
      int m = r + hf * 8;
      s_h2[m][nt * 16 + l]      = (_Float16)dsilu(a0[r] + bb);
      s_h2[m + 16][nt * 16 + l] = (_Float16)dsilu(a1[r] + bb);
    }
  }
  __syncthreads();
  for (int nt = 0; nt < 8; ++nt) {           // 128 -> 128, + residual
    v8f a0 = {}, a1 = {};
    wmma_pair(&s_h2[0][0], HD, W2t, HD, nt * 16, HD, lane, a0, a1);
    float bb = b2[nt * 16 + l];
    int n = nt * 16 + l;
#pragma unroll
    for (int r = 0; r < 8; ++r) {
      int m = r + hf * 8;
      s_out[m][n]      = h[(long)(node0 + m) * HD + n] + a0[r] + bb;
      s_out[m + 16][n] = h[(long)(node0 + m + 16) * HD + n] + a1[r] + bb;
    }
  }
  __syncthreads();
  {                                           // layernorm: one node per lane
    float mean = 0.0f;
    for (int c = 0; c < HD; ++c) mean += s_out[lane][c];
    mean *= (1.0f / HD);
    float var = 0.0f;
    for (int c = 0; c < HD; ++c) { float d = s_out[lane][c] - mean; var += d * d; }
    var *= (1.0f / HD);
    float inv = __builtin_amdgcn_rsqf(var + 1e-5f);
    float* hp = h + (long)(node0 + lane) * HD;
    _Float16* hp16 = h16 + (long)(node0 + lane) * HD;
    for (int c = 0; c < HD; ++c) {
      float o = (s_out[lane][c] - mean) * inv * lng[c] + lnb[c];
      hp[c] = o;
      hp16[c] = (_Float16)o;
    }
  }
}

// ---------------- final: out = vel * scale(h) ----------------
__global__ void fs1_kernel(const float* __restrict__ h,
                           const float* __restrict__ W1, const float* __restrict__ b1,
                           float* __restrict__ qh) {
  int idx = blockIdx.x * blockDim.x + threadIdx.x;
  if (idx >= NQTOT * HD) return;
  int q = idx >> 7, c = idx & (HD - 1);
  const float* hq = h + (long)q * HD;
  float acc = b1[c];
  for (int k = 0; k < HD; ++k) acc += hq[k] * W1[(long)k * HD + c];
  qh[idx] = dsilu(acc);
}
__global__ void fs2_kernel(const float* __restrict__ qh, const float* __restrict__ vel,
                           const float* __restrict__ W2, const float* __restrict__ b2,
                           float* __restrict__ out) {
  int q = blockIdx.x * blockDim.x + threadIdx.x;
  if (q >= NQTOT) return;
  const float* s = qh + (long)q * HD;
  float scale = b2[0];
  for (int c = 0; c < HD; ++c) scale += s[c] * W2[c];
  out[q * 3 + 0] = vel[q * 3 + 0] * scale;
  out[q * 3 + 1] = vel[q * 3 + 1] * scale;
  out[q * 3 + 2] = vel[q * 3 + 2] * scale;
}

// =====================================================================
extern "C" void kernel_launch(void* const* d_in, const int* in_sizes, int n_in,
                              void* d_out, int out_size, void* d_ws, size_t ws_size,
                              hipStream_t stream) {
  (void)in_sizes; (void)n_in; (void)out_size; (void)ws_size;
  const float* x_t        = (const float*)d_in[0];
  const float* t_query    = (const float*)d_in[1];
  const float* atom_pos   = (const float*)d_in[2];
  const float* atom_feat  = (const float*)d_in[3];
  const int*   eaq_src    = (const int*)d_in[4];
  const int*   eaq_tgt    = (const int*)d_in[5];
  const int*   eqq_src    = (const int*)d_in[6];
  const int*   eqq_tgt    = (const int*)d_in[7];
  const float* atom_proj_W = (const float*)d_in[8];
  const float* atom_proj_b = (const float*)d_in[9];
  const float* time_W1 = (const float*)d_in[10];
  const float* time_b1 = (const float*)d_in[11];
  const float* time_W2 = (const float*)d_in[12];
  const float* time_b2 = (const float*)d_in[13];
  const float* qi_W1 = (const float*)d_in[14];
  const float* qi_b1 = (const float*)d_in[15];
  const float* qi_W2 = (const float*)d_in[16];
  const float* qi_b2 = (const float*)d_in[17];
  const float* msg_W1 = (const float*)d_in[18];
  const float* msg_b1 = (const float*)d_in[19];
  const float* msg_W2 = (const float*)d_in[20];
  const float* msg_b2 = (const float*)d_in[21];
  const float* cw_W1 = (const float*)d_in[22];
  const float* cw_b1 = (const float*)d_in[23];
  const float* cw_W2 = (const float*)d_in[24];
  const float* cw_b2 = (const float*)d_in[25];
  const float* nu_W1 = (const float*)d_in[26];
  const float* nu_b1 = (const float*)d_in[27];
  const float* nu_W2 = (const float*)d_in[28];
  const float* nu_b2 = (const float*)d_in[29];
  const float* ln_g = (const float*)d_in[30];
  const float* ln_b = (const float*)d_in[31];
  const float* vo_W1 = (const float*)d_in[32];
  const float* vo_b1 = (const float*)d_in[33];
  const float* vo_W2 = (const float*)d_in[34];
  const float* vo_b2 = (const float*)d_in[35];

  // ---- workspace carve-out ----
  char* ws = (char*)d_ws;
  size_t off = 0;
  auto carve = [&](size_t bytes) -> void* {
    void* p = ws + off;
    off = (off + bytes + 255) & ~(size_t)255;
    return p;
  };
  _Float16* hatom16 = (_Float16*)carve((size_t)NATOT * HD * 2);
  _Float16* temb16  = (_Float16*)carve((size_t)NQTOT * HD * 2);
  _Float16* h16     = (_Float16*)carve((size_t)NQTOT * HD * 2);
  float* h      = (float*)carve((size_t)NQTOT * HD * 4);
  float* agg    = (float*)carve((size_t)NQTOT * HD * 4);
  float* vel    = (float*)carve((size_t)NQTOT * 3 * 4);
  float* ld     = (float*)carve((size_t)NQTOT * 4);
  float* qhid   = (float*)carve((size_t)NQTOT * HD * 4);
  _Float16* timeW1t = (_Float16*)carve((size_t)512 * HD * 2);
  _Float16* timeW2t = (_Float16*)carve((size_t)HD * 512 * 2);
  _Float16 *msgW1t[8], *msgW2t[8], *cwW1t[8], *nuW1t[8], *nuW2t[8];
  for (int j = 0; j < 8; ++j) {
    msgW1t[j] = (_Float16*)carve((size_t)256 * MIKP * 2);
    msgW2t[j] = (_Float16*)carve((size_t)HD * 256 * 2);
    cwW1t[j]  = (_Float16*)carve((size_t)64 * HD * 2);
    nuW1t[j]  = (_Float16*)carve((size_t)HD * 256 * 2);
    nuW2t[j]  = (_Float16*)carve((size_t)HD * HD * 2);
  }

  // ---- weight prep ----
  auto tr = [&](const float* W, _Float16* Wt, int K, int N, int Kp) {
    int total = N * Kp;
    transpose_f16_kernel<<<(total + 255) / 256, 256, 0, stream>>>(W, Wt, K, N, Kp);
  };
  tr(time_W1, timeW1t, HD, 512, HD);
  tr(time_W2, timeW2t, 512, HD, 512);
  for (int j = 0; j < 8; ++j) {
    transpose_msgW1_kernel<<<(256 * MIKP + 255) / 256, 256, 0, stream>>>(
        msg_W1 + (size_t)j * MIK * 256, msgW1t[j]);
    tr(msg_W2 + (size_t)j * 256 * HD,  msgW2t[j], 256, HD, 256);
    tr(cw_W1  + (size_t)j * HD * 64,   cwW1t[j],  HD, 64, HD);
    tr(nu_W1  + (size_t)j * 256 * HD,  nuW1t[j],  256, HD, 256);
    tr(nu_W2  + (size_t)j * HD * HD,   nuW2t[j],  HD, HD, HD);
  }

  // ---- preamble ----
  atom_proj_kernel<<<(NATOT * HD + 255) / 256, 256, 0, stream>>>(
      atom_feat, atom_proj_W, atom_proj_b, hatom16);
  time_kernel<<<NQTOT / 32, 32, 0, stream>>>(
      t_query, timeW1t, time_b1, timeW2t, time_b2, temb16);
  qdist_kernel<<<NQTOT / 256, 256, 0, stream>>>(x_t, atom_pos, ld);
  qhid_kernel<<<NQTOT * HD / 256, 256, 0, stream>>>(ld, qi_W1, qi_b1, qhid);
  dense128_kernel<<<NQTOT * HD / 256, 256, 0, stream>>>(qhid, qi_W2, qi_b2, h, h16);
  hipMemsetAsync(vel, 0, (size_t)NQTOT * 3 * 4, stream);

  // ---- 4 layers x (atom->query, query->query) ----
  for (int i = 0; i < 4; ++i) {
    int j = 2 * i;
    hipMemsetAsync(agg, 0, (size_t)NQTOT * HD * 4, stream);
    edge_kernel<<<EAQN / 32, 32, 0, stream>>>(
        eaq_src, eaq_tgt, hatom16, h16, atom_pos, x_t, temb16,
        msgW1t[j], msg_b1 + (size_t)j * 256, msgW2t[j], msg_b2 + (size_t)j * HD,
        cwW1t[j], cw_b1 + (size_t)j * 64, cw_W2 + (size_t)j * 64, cw_b2 + j,
        agg, vel);
    node_kernel<<<NQTOT / 32, 32, 0, stream>>>(
        h, h16, agg, nuW1t[j], nu_b1 + (size_t)j * HD, nuW2t[j], nu_b2 + (size_t)j * HD,
        ln_g + (size_t)j * HD, ln_b + (size_t)j * HD);

    j = 2 * i + 1;
    hipMemsetAsync(agg, 0, (size_t)NQTOT * HD * 4, stream);
    edge_kernel<<<EQQN / 32, 32, 0, stream>>>(
        eqq_src, eqq_tgt, h16, h16, x_t, x_t, temb16,
        msgW1t[j], msg_b1 + (size_t)j * 256, msgW2t[j], msg_b2 + (size_t)j * HD,
        cwW1t[j], cw_b1 + (size_t)j * 64, cw_W2 + (size_t)j * 64, cw_b2 + j,
        agg, vel);
    node_kernel<<<NQTOT / 32, 32, 0, stream>>>(
        h, h16, agg, nuW1t[j], nu_b1 + (size_t)j * HD, nuW2t[j], nu_b2 + (size_t)j * HD,
        ln_g + (size_t)j * HD, ln_b + (size_t)j * HD);
  }

  // ---- output ----
  fs1_kernel<<<NQTOT * HD / 256, 256, 0, stream>>>(h, vo_W1, vo_b1, qhid);
  fs2_kernel<<<NQTOT / 256, 256, 0, stream>>>(qhid, vel, vo_W2, vo_b2, (float*)d_out);
}
